// Encoder3_45913200394641
// MI455X (gfx1250) — matrix operand order repaired and checked by reading, not compiled, not run on hardware
//
#include <hip/hip_runtime.h>
#include <stdint.h>

#define HID 128
#define STRIPS_PER_BLOCK 10   // 6250 strips of 16 rows / 10 = 625 blocks

typedef __attribute__((ext_vector_type(16))) _Float16 v16h;
typedef __attribute__((ext_vector_type(8)))  _Float16 v8h;
typedef __attribute__((ext_vector_type(8)))  float    v8f;

// ---------------------------------------------------------------- utilities
__device__ __forceinline__ uint32_t hash_u32(uint32_t x) {
    x ^= x >> 16; x *= 0x7feb352dU;
    x ^= x >> 15; x *= 0x846ca68bU;
    x ^= x >> 16;
    return x;
}

// ---------------------------------------------------------------- degree / norm
__global__ void k_deg_init(float* __restrict__ deg, int n) {
    int i = blockIdx.x * blockDim.x + threadIdx.x;
    if (i < n) deg[i] = 1.0f;                       // self-loop contributes 1
}

__global__ void k_deg_accum(const long long* __restrict__ col,
                            float* __restrict__ deg, int n_edges) {
    int e = blockIdx.x * blockDim.x + threadIdx.x;
    if (e < n_edges) atomicAdd(&deg[(int)col[e]], 1.0f);
}

__global__ void k_dinv(float* __restrict__ deg, int n) {
    int i = blockIdx.x * blockDim.x + threadIdx.x;
    if (i < n) deg[i] = rsqrtf(deg[i]);             // deg >= 1 always
}

// ---------------------------------------------------------------- WMMA GEMM
// C[n_rows x 128] = A[n_rows x 128] * W[128 x 128]
// Block: 256 threads = 8 waves. Block stages W^T once (32 KB f16 LDS), then
// iterates STRIPS_PER_BLOCK strips of 16 rows with a double-buffered A strip
// (one workgroup barrier per strip). Wave w owns output cols [16w, 16w+16).
__global__ __launch_bounds__(256) void k_gemm_wmma(
        const float* __restrict__ A, const float* __restrict__ W,
        float* __restrict__ C, int n_strips) {
    __shared__ alignas(16) _Float16 w_lds[HID * HID];     // [n][k]  32 KB
    __shared__ alignas(16) _Float16 a_lds[2][16 * HID];   // [m][k]   8 KB ping-pong

    const int t = threadIdx.x;

    // stage W transposed, f32 -> f16 (once per block)
    #pragma unroll
    for (int i = 0; i < (HID * HID) / 256; ++i) {
        int idx = i * 256 + t;                // coalesced global read
        int k = idx >> 7, n = idx & 127;
        w_lds[n * HID + k] = (_Float16)W[idx];
    }

    const int wv   = t >> 5;        // wave id 0..7 -> column tile
    const int lane = t & 31;
    const int half = lane >> 4;     // 0: lanes 0-15, 1: lanes 16-31
    const int lm   = lane & 15;     // A: M-row / B: N-col / C: N-col

    const int s0 = blockIdx.x * STRIPS_PER_BLOCK;
    int sEnd = s0 + STRIPS_PER_BLOCK;
    if (sEnd > n_strips) sEnd = n_strips;

    for (int s = s0, p = 0; s < sEnd; ++s, p ^= 1) {
        const int row0 = s * 16;
        // stage 16-row A strip into buffer p (safe: barrier of previous
        // iteration guarantees compute on buffer p is finished block-wide)
        #pragma unroll
        for (int i = 0; i < (16 * HID) / 256; ++i) {
            int idx = i * 256 + t;
            a_lds[p][idx] = (_Float16)A[(size_t)row0 * HID + idx];
        }
        __syncthreads();

        v8f c = {};
        #pragma unroll
        for (int k0 = 0; k0 < HID; k0 += 32) {
            // A fragment (16x32 f16): element e -> K = k0 + (e>>3)*16 + half*8 + (e&7)
            v8h alo = *(const v8h*)&a_lds[p][lm * HID + k0 + half * 8];
            v8h ahi = *(const v8h*)&a_lds[p][lm * HID + k0 + 16 + half * 8];
            v16h a = __builtin_shufflevector(alo, ahi,
                     0,1,2,3,4,5,6,7,8,9,10,11,12,13,14,15);
            // B fragment (32x16 f16): element e -> K = k0 + (e>>3)*16 + half*8 + (e&7)
            const int bb = (wv * 16 + lm) * HID + k0 + half * 8;
            v8h blo = *(const v8h*)&w_lds[bb];
            v8h bhi = *(const v8h*)&w_lds[bb + 16];
            v16h b = __builtin_shufflevector(blo, bhi,
                     0,1,2,3,4,5,6,7,8,9,10,11,12,13,14,15);
            c = __builtin_amdgcn_wmma_f32_16x16x32_f16(
                    /*neg_a=*/false, a, /*neg_b=*/false, b,
                    /*c_mod=*/(short)0, c, /*reuse_a=*/false, /*reuse_b=*/false);
        }

        // C/D layout: VGPR r -> M = r + 8*half, N = lm
        #pragma unroll
        for (int r = 0; r < 8; ++r) {
            int m = r + half * 8;
            C[(size_t)(row0 + m) * HID + wv * 16 + lm] = c[r];
        }
    }
}

// ---------------------------------------------------------------- self-loop init
// agg[i,:] = dinv[i]^2 * xw[i,:]
__global__ void k_selfloop(const float* __restrict__ dinv,
                           const float* __restrict__ xw,
                           float* __restrict__ agg, int total) {
    int idx = blockIdx.x * blockDim.x + threadIdx.x;
    if (idx < total) {
        int i = idx >> 7;
        float d = dinv[i];
        agg[idx] = d * d * xw[idx];
    }
}

// ---------------------------------------------------------------- edge scatter
// Each wave processes 32 edges per chunk: indices + norms loaded coalesced
// (one edge per lane), then each edge is broadcast via wave32 shuffles while
// all 32 lanes move its 128 features (float4/lane gather + 4 f32 atomics).
__global__ __launch_bounds__(256) void k_edge_scatter(
        const long long* __restrict__ row, const long long* __restrict__ col,
        const float* __restrict__ dinv, const float* __restrict__ xw,
        float* __restrict__ agg, int n_edges) {
    const int lane = threadIdx.x & 31;
    const int wid  = threadIdx.x >> 5;
    const int stride = gridDim.x * 8 * 32;

    for (int base = (blockIdx.x * 8 + wid) * 32; base < n_edges; base += stride) {
        int e = base + lane;
        int r = 0, c = 0;
        float nrm = 0.0f;
        if (e < n_edges) {
            r = (int)row[e];                 // coalesced int64 loads
            c = (int)col[e];
            nrm = dinv[r] * dinv[c];
        }
        int cnt = n_edges - base;
        if (cnt > 32) cnt = 32;

        for (int j = 0; j < cnt; ++j) {
            int   rj = __shfl(r, j, 32);     // broadcast edge j to all lanes
            int   cj = __shfl(c, j, 32);
            float nj = __shfl(nrm, j, 32);
            if (j + 1 < cnt) {               // prefetch next edge's source row
                int rn = __shfl(r, j + 1, 32);
                __builtin_prefetch(xw + (size_t)rn * HID + lane * 4, 0, 0);
            }
            float4 v = ((const float4*)(xw + (size_t)rj * HID))[lane];
            float* dst = agg + (size_t)cj * HID + lane * 4;
            atomicAdd(dst + 0, nj * v.x);
            atomicAdd(dst + 1, nj * v.y);
            atomicAdd(dst + 2, nj * v.z);
            atomicAdd(dst + 3, nj * v.w);
        }
    }
}

// ---------------------------------------------------------------- bias+dropout+prelu
__global__ void k_post(const float* __restrict__ agg, const float* __restrict__ bias,
                       const float* __restrict__ alpha, float* __restrict__ out,
                       int total, uint32_t seed) {
    int idx = blockIdx.x * blockDim.x + threadIdx.x;
    if (idx < total) {
        int f = idx & 127;
        float h = agg[idx] + bias[f];
        // deterministic hash-based dropout, p = 0.1
        uint32_t u = hash_u32((uint32_t)idx ^ seed);
        h = (u >= 429496730u) ? h * (1.0f / 0.9f) : 0.0f;
        out[idx] = (h >= 0.0f) ? h : alpha[f] * h;
    }
}

// ---------------------------------------------------------------- launcher
extern "C" void kernel_launch(void* const* d_in, const int* in_sizes, int n_in,
                              void* d_out, int out_size, void* d_ws, size_t ws_size,
                              hipStream_t stream) {
    (void)n_in; (void)out_size; (void)ws_size;
    const float*     x     = (const float*)d_in[0];
    const long long* ei    = (const long long*)d_in[1];   // int64 [2, E]
    const float*     W1    = (const float*)d_in[2];
    const float*     b1    = (const float*)d_in[3];
    const float*     W2    = (const float*)d_in[4];
    const float*     b2    = (const float*)d_in[5];
    const float*     alpha = (const float*)d_in[6];
    float*           out   = (float*)d_out;

    const int n_nodes = in_sizes[0] / HID;     // 100000
    const int n_edges = in_sizes[1] / 2;       // 1600000
    const long long* rowp = ei;
    const long long* colp = ei + n_edges;

    // workspace layout: [dinv (n f32)] [bufA 51.2MB] [bufB 51.2MB]  (~103 MB)
    char* ws = (char*)d_ws;
    size_t dinv_bytes = (((size_t)n_nodes * sizeof(float)) + 255) & ~(size_t)255;
    size_t feat_count = (size_t)n_nodes * HID;
    float* dinv = (float*)ws;
    float* bufA = (float*)(ws + dinv_bytes);                               // xw / xw2
    float* bufB = (float*)(ws + dinv_bytes + feat_count * sizeof(float));  // agg / h1

    const int total    = (int)feat_count;
    const int n_strips = n_nodes / 16;         // 100000 % 16 == 0 -> 6250
    const int gN       = (n_nodes + 255) / 256;
    const int gE       = (n_edges + 255) / 256;
    const int gF       = (total + 255) / 256;
    const int gGemm    = (n_strips + STRIPS_PER_BLOCK - 1) / STRIPS_PER_BLOCK; // 625
    const int gEdge    = (n_edges + 8 * 32 - 1) / (8 * 32); // 32 edges/wave -> 6250

    // normalization: dinv = rsqrt(1 + in-degree)
    k_deg_init <<<gN, 256, 0, stream>>>(dinv, n_nodes);
    k_deg_accum<<<gE, 256, 0, stream>>>(colp, dinv, n_edges);
    k_dinv     <<<gN, 256, 0, stream>>>(dinv, n_nodes);

    // ---- layer 1
    k_gemm_wmma   <<<gGemm, 256, 0, stream>>>(x, W1, bufA, n_strips);          // xw1
    k_selfloop    <<<gF,    256, 0, stream>>>(dinv, bufA, bufB, total);        // agg init
    k_edge_scatter<<<gEdge, 256, 0, stream>>>(rowp, colp, dinv, bufA, bufB, n_edges);
    k_post        <<<gF,    256, 0, stream>>>(bufB, b1, alpha, bufB, total, 0x9e3779b9u); // h1 in-place

    // ---- layer 2
    k_gemm_wmma   <<<gGemm, 256, 0, stream>>>(bufB, W2, bufA, n_strips);       // xw2
    k_selfloop    <<<gF,    256, 0, stream>>>(dinv, bufA, bufB, total);        // agg2 (h1 dead)
    k_edge_scatter<<<gEdge, 256, 0, stream>>>(rowp, colp, dinv, bufA, bufB, n_edges);
    k_post        <<<gF,    256, 0, stream>>>(bufB, b2, alpha, out, total, 0x85ebca6bu);
}
